// CLUB_1159641170012
// MI455X (gfx1250) — compile-verified, matrix operand order checked by reading, not run on hardware
//
#include <hip/hip_runtime.h>
#include <hip/hip_bf16.h>

#define BATCH 8192
#define HDIM  768

typedef __attribute__((ext_vector_type(16))) _Float16 v16h;
typedef __attribute__((ext_vector_type(8)))  _Float16 v8h;
typedef __attribute__((ext_vector_type(4)))  _Float16 v4h;
typedef __attribute__((ext_vector_type(8)))  float    v8f;
typedef __attribute__((ext_vector_type(4)))  float    v4f;

// ---------------------------------------------------------------------------
// D = A(16x32 f16) x B(32x16 f16) + C(16x16 f32), wave32 WMMA
// ---------------------------------------------------------------------------
__device__ __forceinline__ v8f wmma16(v16h a, v16h b, v8f c) {
  return __builtin_amdgcn_wmma_f32_16x16x32_f16(
      /*neg_a=*/false, a, /*neg_b=*/false, b,
      /*c_mod=*/(short)0, c, /*reuse_a=*/false, /*reuse_b=*/false);
}

// Per-lane fixed element offset of a fragment stream. lane l -> row
// (row + (l&15)), element offset koff (8 for lanes 16..31).
__device__ __forceinline__ unsigned frag_off(int row, int lane) {
  return (unsigned)((row + (lane & 15)) * HDIM + ((lane >> 4) << 3));
}

// p is a per-stream base pointer hoisted out of the (fully unrolled) k-loop;
// k0 is a compile-time constant, so both b128 loads use the SAME base VGPR
// pair with immediate offsets -> zero address VALU in the main loop.
__device__ __forceinline__ v16h load_frag(const _Float16* __restrict__ p, int k0) {
  union { v16h v; v8h h[2]; } u;
  u.h[0] = *(const v8h*)(p + k0);        // K = koff .. koff+7
  u.h[1] = *(const v8h*)(p + k0 + 16);   // K = 16+koff .. 16+koff+7
  return u.v;
}

// ---------------------------------------------------------------------------
// fp32 -> fp16 cast, 4 elements/thread (b128 in, b64 out)
// ---------------------------------------------------------------------------
__global__ __launch_bounds__(256) void cvt_f32_f16(const float* __restrict__ in,
                                                   _Float16* __restrict__ out, int n4) {
  int i = blockIdx.x * blockDim.x + threadIdx.x;
  if (i < n4) {
    v4f x = ((const v4f*)in)[i];
    v4h y;
    y.x = (_Float16)x.x; y.y = (_Float16)x.y;
    y.z = (_Float16)x.z; y.w = (_Float16)x.w;
    ((v4h*)out)[i] = y;
  }
}

// ---------------------------------------------------------------------------
// GEMM wave tile: 32(M) x 64(N) for BOTH MLP paths simultaneously.
// Block = 8 waves arranged 4(M) x 2(N)  ->  128 x 128 block tile.
// Per k-step: 2 (layer 1) / 4 (layer 2) A-frag loads + 8 B-frag loads (20 / 24 b128) feed 16 WMMAs.
// k-loop fully unrolled; one base pointer per stream, k0 in load immediates.
// ---------------------------------------------------------------------------

// Layer 1: Hm = tanh(A @ W1m^T + b1m), Hv = tanh(A @ W1v^T + b1v)   [f16 out]
__global__ __launch_bounds__(256) void layer1_kernel(
    const _Float16* __restrict__ A,
    const _Float16* __restrict__ W1m, const _Float16* __restrict__ W1v,
    const float* __restrict__ b1m,    const float* __restrict__ b1v,
    _Float16* __restrict__ Hm,        _Float16* __restrict__ Hv) {
  const int lane = threadIdx.x & 31;
  const int wave = threadIdx.x >> 5;
  const int wm   = wave & 3;
  const int wn   = wave >> 2;
  const int row  = (blockIdx.x * 4 + wm) * 32;       // 32 M-rows per wave
  const int ncol = (blockIdx.y * 2 + wn) * 64;       // 64 N-cols per wave

  // One 64-bit base pointer per fragment stream, hoisted for the whole loop.
  const _Float16* __restrict__ pA0 = A + frag_off(row,      lane);
  const _Float16* __restrict__ pA1 = A + frag_off(row + 16, lane);
  const _Float16* __restrict__ pBm[4];
  const _Float16* __restrict__ pBv[4];
#pragma unroll
  for (int j = 0; j < 4; ++j) {
    const unsigned o = frag_off(ncol + j * 16, lane);
    pBm[j] = W1m + o;
    pBv[j] = W1v + o;
  }

  v8f accm[2][4] = {};
  v8f accv[2][4] = {};

#pragma unroll
  for (int k0 = 0; k0 < HDIM; k0 += 32) {
    v16h a0 = load_frag(pA0, k0);
    v16h a1 = load_frag(pA1, k0);
    v16h bm[4], bv[4];
#pragma unroll
    for (int j = 0; j < 4; ++j) bm[j] = load_frag(pBm[j], k0);
#pragma unroll
    for (int j = 0; j < 4; ++j) bv[j] = load_frag(pBv[j], k0);
#pragma unroll
    for (int j = 0; j < 4; ++j) {
      accm[0][j] = wmma16(a0, bm[j], accm[0][j]);
      accm[1][j] = wmma16(a1, bm[j], accm[1][j]);
      accv[0][j] = wmma16(a0, bv[j], accv[0][j]);
      accv[1][j] = wmma16(a1, bv[j], accv[1][j]);
    }
  }

  // Epilogue: bias + tanh + f16 store. C/D layout: N=lane&15, M=r+8*(lane>>4).
  const int n = lane & 15;
#pragma unroll
  for (int mi = 0; mi < 2; ++mi) {
    const int mbase = row + mi * 16 + ((lane >> 4) << 3);
#pragma unroll
    for (int j = 0; j < 4; ++j) {
      const int col = ncol + j * 16 + n;
      const float bm = b1m[col];
      const float bv = b1v[col];
#pragma unroll
      for (int r = 0; r < 8; ++r) {
        const size_t idx = (size_t)(mbase + r) * HDIM + col;
        Hm[idx] = (_Float16)tanhf(accm[mi][j][r] + bm);
        Hv[idx] = (_Float16)tanhf(accv[mi][j][r] + bv);
      }
    }
  }
}

// Layer 2: mu = Hm @ W2m^T + b2m  [f32],  invvar = exp(-tanh(Hv @ W2v^T + b2v)) [f32]
// Also accumulates per-feature column sums of mu and mu^2 (f32 atomics).
__global__ __launch_bounds__(256) void layer2_kernel(
    const _Float16* __restrict__ Hm,  const _Float16* __restrict__ Hv,
    const _Float16* __restrict__ W2m, const _Float16* __restrict__ W2v,
    const float* __restrict__ b2m,    const float* __restrict__ b2v,
    float* __restrict__ mu,           float* __restrict__ invvar,
    float* __restrict__ colsum_mu,    float* __restrict__ colsum_musq) {
  const int lane = threadIdx.x & 31;
  const int wave = threadIdx.x >> 5;
  const int wm   = wave & 3;
  const int wn   = wave >> 2;
  const int row  = (blockIdx.x * 4 + wm) * 32;
  const int ncol = (blockIdx.y * 2 + wn) * 64;

  const unsigned oA0 = frag_off(row,      lane);
  const unsigned oA1 = frag_off(row + 16, lane);
  const _Float16* __restrict__ pA0 = Hm + oA0;
  const _Float16* __restrict__ pA1 = Hm + oA1;
  const _Float16* __restrict__ pC0 = Hv + oA0;
  const _Float16* __restrict__ pC1 = Hv + oA1;
  const _Float16* __restrict__ pBm[4];
  const _Float16* __restrict__ pBv[4];
#pragma unroll
  for (int j = 0; j < 4; ++j) {
    const unsigned o = frag_off(ncol + j * 16, lane);
    pBm[j] = W2m + o;
    pBv[j] = W2v + o;
  }

  v8f accm[2][4] = {};
  v8f accv[2][4] = {};

#pragma unroll
  for (int k0 = 0; k0 < HDIM; k0 += 32) {
    v16h a0 = load_frag(pA0, k0);
    v16h a1 = load_frag(pA1, k0);
    v16h c0 = load_frag(pC0, k0);
    v16h c1 = load_frag(pC1, k0);
    v16h bm[4], bv[4];
#pragma unroll
    for (int j = 0; j < 4; ++j) bm[j] = load_frag(pBm[j], k0);
#pragma unroll
    for (int j = 0; j < 4; ++j) bv[j] = load_frag(pBv[j], k0);
#pragma unroll
    for (int j = 0; j < 4; ++j) {
      accm[0][j] = wmma16(a0, bm[j], accm[0][j]);
      accm[1][j] = wmma16(a1, bm[j], accm[1][j]);
      accv[0][j] = wmma16(c0, bv[j], accv[0][j]);
      accv[1][j] = wmma16(c1, bv[j], accv[1][j]);
    }
  }

  const int n = lane & 15;
#pragma unroll
  for (int mi = 0; mi < 2; ++mi) {
    const int mbase = row + mi * 16 + ((lane >> 4) << 3);
#pragma unroll
    for (int j = 0; j < 4; ++j) {
      const int col = ncol + j * 16 + n;
      const float bm = b2m[col];
      const float bv = b2v[col];
      float s1 = 0.0f, s2 = 0.0f;
#pragma unroll
      for (int r = 0; r < 8; ++r) {
        const size_t idx = (size_t)(mbase + r) * HDIM + col;
        const float m  = accm[mi][j][r] + bm;
        const float lv = tanhf(accv[mi][j][r] + bv);
        mu[idx]     = m;
        invvar[idx] = __expf(-lv);
        s1 += m;
        s2 += m * m;
      }
      atomicAdd(&colsum_mu[col],   s1);   // lanes l and l+16 hit the same col
      atomicAdd(&colsum_musq[col], s2);
    }
  }
}

// ---------------------------------------------------------------------------
// Final reduction: one block per batch row.
//   positive = -(mu-b)^2 * 0.5 * iv
//   pair_sq  = E[mu^2] - 2*mu*E[mu] + mu^2 ; negative = -pair_sq*0.5*iv
//   out[0] = mean_rows(sum_c positive) ; out[1] = mean_rows(pos - neg)
// ---------------------------------------------------------------------------
__global__ __launch_bounds__(256) void reduce_kernel(
    const float* __restrict__ mu, const float* __restrict__ invvar,
    const float* __restrict__ modal_b,
    const float* __restrict__ colsum_mu, const float* __restrict__ colsum_musq,
    float* __restrict__ out) {
  const int rowid = blockIdx.x;
  const int tid   = threadIdx.x;
  const float invB = 1.0f / (float)BATCH;

  float pos = 0.0f, neg = 0.0f;
  for (int c = tid; c < HDIM; c += 256) {
    const size_t idx = (size_t)rowid * HDIM + c;
    const float m  = mu[idx];
    const float iv = invvar[idx];
    const float b  = modal_b[idx];
    const float mm = colsum_mu[c]   * invB;   // E[mu]    per feature
    const float ms = colsum_musq[c] * invB;   // E[mu^2]  per feature
    const float d  = m - b;
    pos += -d * d * 0.5f * iv;
    const float pair = ms - 2.0f * m * mm + m * m;
    neg += -pair * 0.5f * iv;
  }

  __shared__ float sp[256];
  __shared__ float sn[256];
  sp[tid] = pos;
  sn[tid] = neg;
  __syncthreads();
  for (int s = 128; s > 0; s >>= 1) {
    if (tid < s) { sp[tid] += sp[tid + s]; sn[tid] += sn[tid + s]; }
    __syncthreads();
  }
  if (tid == 0) {
    atomicAdd(&out[0], sp[0] * invB);            // lld
    atomicAdd(&out[1], (sp[0] - sn[0]) * invB);  // bound
  }
}

// ---------------------------------------------------------------------------
// Host-side orchestration
// ---------------------------------------------------------------------------
extern "C" void kernel_launch(void* const* d_in, const int* in_sizes, int n_in,
                              void* d_out, int out_size, void* d_ws, size_t ws_size,
                              hipStream_t stream) {
  (void)in_sizes; (void)n_in; (void)out_size; (void)ws_size;

  const float* modal_a = (const float*)d_in[0];
  const float* modal_b = (const float*)d_in[1];
  const float* W1m = (const float*)d_in[2];
  const float* b1m = (const float*)d_in[3];
  const float* W2m = (const float*)d_in[4];
  const float* b2m = (const float*)d_in[5];
  const float* W1v = (const float*)d_in[6];
  const float* b1v = (const float*)d_in[7];
  const float* W2v = (const float*)d_in[8];
  const float* b2v = (const float*)d_in[9];
  float* out = (float*)d_out;

  const size_t BH = (size_t)BATCH * HDIM;           // 6,291,456
  const size_t WW = (size_t)HDIM * HDIM;            //   589,824
  auto align256 = [](size_t x) { return (x + 255) & ~(size_t)255; };

  char* ws = (char*)d_ws;
  size_t off = 0;
  _Float16* Ah    = (_Float16*)(ws + off); off += align256(BH * 2);
  _Float16* W1m_h = (_Float16*)(ws + off); off += align256(WW * 2);
  _Float16* W2m_h = (_Float16*)(ws + off); off += align256(WW * 2);
  _Float16* W1v_h = (_Float16*)(ws + off); off += align256(WW * 2);
  _Float16* W2v_h = (_Float16*)(ws + off); off += align256(WW * 2);
  _Float16* Hm    = (_Float16*)(ws + off); off += align256(BH * 2);
  _Float16* Hv    = (_Float16*)(ws + off); off += align256(BH * 2);
  float*    mu    = (float*)   (ws + off); off += align256(BH * 4);
  float*    iv    = (float*)   (ws + off); off += align256(BH * 4);
  float*    cs    = (float*)   (ws + off); off += align256(2 * HDIM * 4);
  float*    cs_mu   = cs;
  float*    cs_musq = cs + HDIM;

  // Zero accumulation targets (graph-capture safe).
  hipMemsetAsync(cs, 0, 2 * HDIM * sizeof(float), stream);
  hipMemsetAsync(out, 0, 2 * sizeof(float), stream);

  // fp32 -> fp16 casts (A once, 4 weight matrices), 4 elems/thread.
  cvt_f32_f16<<<dim3((BH / 4 + 255) / 256), dim3(256), 0, stream>>>(modal_a, Ah, (int)(BH / 4));
  cvt_f32_f16<<<dim3((WW / 4 + 255) / 256), dim3(256), 0, stream>>>(W1m, W1m_h, (int)(WW / 4));
  cvt_f32_f16<<<dim3((WW / 4 + 255) / 256), dim3(256), 0, stream>>>(W2m, W2m_h, (int)(WW / 4));
  cvt_f32_f16<<<dim3((WW / 4 + 255) / 256), dim3(256), 0, stream>>>(W1v, W1v_h, (int)(WW / 4));
  cvt_f32_f16<<<dim3((WW / 4 + 255) / 256), dim3(256), 0, stream>>>(W2v, W2v_h, (int)(WW / 4));

  // GEMM grids: 128x128 block tile -> (8192/128, 768/128) = (64, 6).
  dim3 ggrid(BATCH / 128, HDIM / 128);
  dim3 gblock(256);
  layer1_kernel<<<ggrid, gblock, 0, stream>>>(Ah, W1m_h, W1v_h, b1m, b1v, Hm, Hv);
  layer2_kernel<<<ggrid, gblock, 0, stream>>>(Hm, Hv, W2m_h, W2v_h, b2m, b2v,
                                              mu, iv, cs_mu, cs_musq);

  // Final scalar reduction: one block per batch row.
  reduce_kernel<<<dim3(BATCH), dim3(256), 0, stream>>>(mu, iv, modal_b,
                                                       cs_mu, cs_musq, out);
}
